// QuanvolutionFilter_65481071405125
// MI455X (gfx1250) — matrix operand order repaired and checked by reading, not compiled, not run on hardware
//
#include <hip/hip_runtime.h>

typedef __attribute__((ext_vector_type(16))) _Float16     v16h;
typedef __attribute__((ext_vector_type(8)))  float        v8f;
typedef __attribute__((ext_vector_type(8)))  unsigned int v8u;

#define N_LAYERS 3

__global__ __launch_bounds__(256) void quanv_wmma_kernel(
    const float* __restrict__ x,    // [4096, 784] (28x28 images, flat)
    const float* __restrict__ wts,  // [3, 4]
    float* __restrict__ out,        // [totalPatches, 4]
    unsigned totalPatches)
{
    const int tid  = threadIdx.x;
    const int lane = tid & 31;

    // ---- patch addressing (one lane = one patch) ----
    const unsigned P  = blockIdx.x * 256u + (unsigned)tid;
    const unsigned Pc = (P < totalPatches) ? P : (totalPatches - 1u);  // keep EXEC full
    const unsigned img = Pc / 196u;
    const unsigned q   = Pc - img * 196u;
    const unsigned r   = q / 14u;
    const unsigned c   = q - r * 14u;

    // coalesced: 14 consecutive lanes cover one contiguous 112B row segment
    const float* px = x + (size_t)img * 784u + (2u * r) * 28u + 2u * c;
    const float2 row0 = *(const float2*)(px);
    const float2 row1 = *(const float2*)(px + 28);
    float pv[4] = { row0.x, row0.y, row1.x, row1.y };

    // weights: uniform -> scalar loads
    float wt[N_LAYERS][4];
    #pragma unroll
    for (int l = 0; l < N_LAYERS; ++l)
        #pragma unroll
        for (int w = 0; w < 4; ++w)
            wt[l][w] = wts[l * 4 + w];

    // ---- 4-qubit statevector simulation, fully register-resident ----
    // state index: bit3=wire0, bit2=wire1, bit1=wire2, bit0=wire3
    float s[16];
    #pragma unroll
    for (int i = 0; i < 16; ++i) s[i] = 0.0f;
    s[0] = 1.0f;

    #pragma unroll
    for (int l = 0; l < N_LAYERS; ++l) {
        #pragma unroll
        for (int w = 0; w < 4; ++w) {
            const float h = 0.5f * pv[w] * wt[l][w];
            float sn, cs;
            __sincosf(h, &sn, &cs);
            const int bit = 8 >> w;
            #pragma unroll
            for (int idx = 0; idx < 16; ++idx) {
                if ((idx & bit) == 0) {
                    const float a0 = s[idx], a1 = s[idx | bit];
                    s[idx]       = cs * a0 - sn * a1;
                    s[idx | bit] = sn * a0 + cs * a1;
                }
            }
        }
        // CNOT(0,1): for wire0==1 flip wire1 -> swap blocks [8..11] <-> [12..15]
        #pragma unroll
        for (int j = 0; j < 4; ++j) { float t = s[8 + j]; s[8 + j] = s[12 + j]; s[12 + j] = t; }
        // CNOT(2,3): for wire2==1 flip wire3 -> swap (k+2, k+3)
        #pragma unroll
        for (int k = 0; k < 16; k += 4) { float t = s[k + 2]; s[k + 2] = s[k + 3]; s[k + 3] = t; }
    }

    // ---- probs in f16 hi/lo split (error ~2^-22: f32-exact contraction) ----
    unsigned pkhi[8], pklo[8];
    #pragma unroll
    for (int v = 0; v < 8; ++v) {
        const float p0 = s[2 * v] * s[2 * v];
        const float p1 = s[2 * v + 1] * s[2 * v + 1];
        const _Float16 h0 = (_Float16)p0, h1 = (_Float16)p1;
        const _Float16 l0 = (_Float16)(p0 - (float)h0);
        const _Float16 l1 = (_Float16)(p1 - (float)h1);
        const unsigned uh0 = (unsigned)__builtin_bit_cast(unsigned short, h0);
        const unsigned uh1 = (unsigned)__builtin_bit_cast(unsigned short, h1);
        const unsigned ul0 = (unsigned)__builtin_bit_cast(unsigned short, l0);
        const unsigned ul1 = (unsigned)__builtin_bit_cast(unsigned short, l1);
        pkhi[v] = uh0 | (uh1 << 16);
        pklo[v] = ul0 | (ul1 << 16);
    }

    // ---- A = sign matrix (16x32 f16 layout): rows m<4, K<16 = +/-1, else 0 ----
    const int m    = lane & 15;
    const int koff = (lane & 16) ? 8 : 0;
    v16h A = {};
    #pragma unroll
    for (int e = 0; e < 16; ++e) {
        const int k = 2 * koff + e;
        const _Float16 sg = ((k >> (3 - (m & 3))) & 1) ? (_Float16)(-1.0f) : (_Float16)(1.0f);
        A[e] = (m < 4 && k < 16) ? sg : (_Float16)0.0f;   // lanes 16-31 (K>=16) stay 0
    }

    // ---- B = prob columns (32x16 f16 layout): lanes 0-15 K=0..15, lanes 16-31 zero ----
    // group0 = this wave's patches 0..15 (own lanes), group1 = patches 16..31 (bpermute +16)
    const int permAddr = ((lane + 16) & 31) << 2;
    v8u bh0, bl0, bh1, bl1;
    #pragma unroll
    for (int v = 0; v < 8; ++v) {
        const unsigned oh = (unsigned)__builtin_amdgcn_ds_bpermute(permAddr, (int)pkhi[v]);
        const unsigned ol = (unsigned)__builtin_amdgcn_ds_bpermute(permAddr, (int)pklo[v]);
        bh0[v] = (lane < 16) ? pkhi[v] : 0u;
        bl0[v] = (lane < 16) ? pklo[v] : 0u;
        bh1[v] = (lane < 16) ? oh : 0u;
        bl1[v] = (lane < 16) ? ol : 0u;
    }
    const v16h Bh0 = __builtin_bit_cast(v16h, bh0);
    const v16h Bl0 = __builtin_bit_cast(v16h, bl0);
    const v16h Bh1 = __builtin_bit_cast(v16h, bh1);
    const v16h Bl1 = __builtin_bit_cast(v16h, bl1);

    // patch values of lane+16 (for group1 epilogue add)
    float pvO[4];
    #pragma unroll
    for (int w = 0; w < 4; ++w)
        pvO[w] = __builtin_bit_cast(float,
                   __builtin_amdgcn_ds_bpermute(permAddr, __builtin_bit_cast(int, pv[w])));

    // ---- D = Sign x Probs, chained hi+lo accumulation in f32 (EXEC all-ones) ----
    v8f c0 = {}, c1 = {};
    c0 = __builtin_amdgcn_wmma_f32_16x16x32_f16(false, A, false, Bh0, (short)0, c0, false, false);
    c0 = __builtin_amdgcn_wmma_f32_16x16x32_f16(false, A, false, Bl0, (short)0, c0, false, false);
    c1 = __builtin_amdgcn_wmma_f32_16x16x32_f16(false, A, false, Bh1, (short)0, c1, false, false);
    c1 = __builtin_amdgcn_wmma_f32_16x16x32_f16(false, A, false, Bl1, (short)0, c1, false, false);

    // ---- store: lanes 0-15 hold wires 0..3 (VGPRs 0..3) for both groups ----
    if (lane < 16) {
        const unsigned waveBase = blockIdx.x * 256u + (unsigned)(tid & ~31);
        const unsigned p0i = waveBase + (unsigned)lane;
        const unsigned p1i = waveBase + 16u + (unsigned)lane;
        if (p0i < totalPatches) {
            float4 o0 = make_float4(c0[0] + pv[0], c0[1] + pv[1], c0[2] + pv[2], c0[3] + pv[3]);
            *(float4*)(out + (size_t)p0i * 4u) = o0;
        }
        if (p1i < totalPatches) {
            float4 o1 = make_float4(c1[0] + pvO[0], c1[1] + pvO[1], c1[2] + pvO[2], c1[3] + pvO[3]);
            *(float4*)(out + (size_t)p1i * 4u) = o1;
        }
    }
}

extern "C" void kernel_launch(void* const* d_in, const int* in_sizes, int n_in,
                              void* d_out, int out_size, void* d_ws, size_t ws_size,
                              hipStream_t stream) {
    (void)n_in; (void)out_size; (void)d_ws; (void)ws_size;
    const float* x  = (const float*)d_in[0];   // [4096,1,28,28] f32
    const float* w  = (const float*)d_in[1];   // [3,4] f32
    float* out      = (float*)d_out;           // [4096, 784] f32

    const int totalPatches = in_sizes[0] / 4;          // 4 pixels per 2x2 patch
    const int grid = (totalPatches + 255) / 256;       // 3136 for 4096 images
    quanv_wmma_kernel<<<grid, 256, 0, stream>>>(x, w, out, (unsigned)totalPatches);
}
